// LLFEncoder_81750407512364
// MI455X (gfx1250) — matrix operand order repaired and checked by reading, not compiled, not run on hardware
//
#include <hip/hip_runtime.h>
#include <math.h>
#include <stdint.h>

// ---------------------------------------------------------------------------
// Types for CDNA5 WMMA (gfx1250, wave32): v_wmma_f32_16x16x32_bf16
// ---------------------------------------------------------------------------
typedef __bf16 v16bf __attribute__((ext_vector_type(16)));
typedef __bf16 v8bf  __attribute__((ext_vector_type(8)));
typedef float  v8f   __attribute__((ext_vector_type(8)));

union Frag16 {           // 16 bf16 = 32 bytes
    v16bf    v;
    v8bf     h[2];
    unsigned u[8];
};

#define EPSB 1e-5f

__device__ __forceinline__ void wait_async0() {
#if __has_builtin(__builtin_amdgcn_s_wait_asynccnt)
    __builtin_amdgcn_s_wait_asynccnt(0);
#else
    asm volatile("s_wait_asynccnt 0" ::: "memory");
#endif
}

// ---------------------------------------------------------------------------
// 1) Bilinear resize mfcc(100,120)->64x128 + llfs(100,96)->64x128, concat
//    into t (16,128,128) fp32.  Half-pixel sampling, clamped.
// ---------------------------------------------------------------------------
__device__ __forceinline__ float bilin_sample(const float* __restrict__ src,
                                              int H, int W, float fy, float fx) {
    int y0 = (int)floorf(fy);
    int x0 = (int)floorf(fx);
    float wy = fy - (float)y0, wx = fx - (float)x0;
    int y0c = min(max(y0, 0), H - 1);
    int y1c = min(max(y0 + 1, 0), H - 1);
    int x0c = min(max(x0, 0), W - 1);
    int x1c = min(max(x0 + 1, 0), W - 1);
    float a = src[y0c * W + x0c], b = src[y0c * W + x1c];
    float c = src[y1c * W + x0c], d = src[y1c * W + x1c];
    float top = a + (b - a) * wx;
    float bot = c + (d - c) * wx;
    return top + (bot - top) * wy;
}

__global__ void resize_concat_kernel(const float* __restrict__ mfcc,
                                     const float* __restrict__ llfs,
                                     float* __restrict__ t) {
    int i = blockIdx.x * blockDim.x + threadIdx.x;      // 16*128*128
    if (i >= 16 * 128 * 128) return;
    int x = i % 128;
    int y = (i / 128) % 128;
    int b = i / (128 * 128);
    float v;
    if (y < 64) {
        float fy = (y + 0.5f) * (100.f / 64.f) - 0.5f;
        float fx = (x + 0.5f) * (120.f / 128.f) - 0.5f;
        v = bilin_sample(mfcc + b * 100 * 120, 100, 120, fy, fx);
    } else {
        float fy = ((y - 64) + 0.5f) * (100.f / 64.f) - 0.5f;
        float fx = (x + 0.5f) * (96.f / 128.f) - 0.5f;
        v = bilin_sample(llfs + b * 100 * 96, 100, 96, fy, fx);
    }
    t[i] = v;
}

// ---------------------------------------------------------------------------
// 2) Global min/max over t (two pass)
// ---------------------------------------------------------------------------
__global__ void minmax_part_kernel(const float* __restrict__ x, int n,
                                   float* __restrict__ part) {
    __shared__ float smn[256], smx[256];
    float mn = 3.4e38f, mx = -3.4e38f;
    for (int i = blockIdx.x * blockDim.x + threadIdx.x; i < n;
         i += gridDim.x * blockDim.x) {
        float v = x[i];
        mn = fminf(mn, v);
        mx = fmaxf(mx, v);
    }
    smn[threadIdx.x] = mn; smx[threadIdx.x] = mx;
    __syncthreads();
    for (int s = 128; s > 0; s >>= 1) {
        if (threadIdx.x < (unsigned)s) {
            smn[threadIdx.x] = fminf(smn[threadIdx.x], smn[threadIdx.x + s]);
            smx[threadIdx.x] = fmaxf(smx[threadIdx.x], smx[threadIdx.x + s]);
        }
        __syncthreads();
    }
    if (threadIdx.x == 0) {
        part[blockIdx.x]       = smn[0];
        part[256 + blockIdx.x] = smx[0];
    }
}

__global__ void minmax_final_kernel(const float* __restrict__ part,
                                    float* __restrict__ mm) {
    __shared__ float smn[256], smx[256];
    smn[threadIdx.x] = part[threadIdx.x];
    smx[threadIdx.x] = part[256 + threadIdx.x];
    __syncthreads();
    for (int s = 128; s > 0; s >>= 1) {
        if (threadIdx.x < (unsigned)s) {
            smn[threadIdx.x] = fminf(smn[threadIdx.x], smn[threadIdx.x + s]);
            smx[threadIdx.x] = fmaxf(smx[threadIdx.x], smx[threadIdx.x + s]);
        }
        __syncthreads();
    }
    if (threadIdx.x == 0) { mm[0] = smn[0]; mm[1] = smx[0]; }
}

// ---------------------------------------------------------------------------
// 3) Normalize -> LUT colormap (256x3) at 128x128, bilinear resize to 224x224
//    fused (gather-then-resize, matching the reference order).
//    Output img fp32 NCHW (16,3,224,224).
// ---------------------------------------------------------------------------
__global__ void colorize_resize_kernel(const float* __restrict__ t,
                                       const float* __restrict__ mm,
                                       const float* __restrict__ lut,
                                       float* __restrict__ img) {
    int i = blockIdx.x * blockDim.x + threadIdx.x;      // 16*224*224
    if (i >= 16 * 224 * 224) return;
    int ox = i % 224;
    int oy = (i / 224) % 224;
    int b  = i / (224 * 224);
    float mn = mm[0], inv = 1.f / (mm[1] - mm[0]);

    float fy = (oy + 0.5f) * (128.f / 224.f) - 0.5f;
    float fx = (ox + 0.5f) * (128.f / 224.f) - 0.5f;
    int y0 = (int)floorf(fy), x0 = (int)floorf(fx);
    float wy = fy - (float)y0, wx = fx - (float)x0;
    int yy[2] = { min(max(y0, 0), 127), min(max(y0 + 1, 0), 127) };
    int xx[2] = { min(max(x0, 0), 127), min(max(x0 + 1, 0), 127) };
    float wgt[2][2] = { { (1.f - wy) * (1.f - wx), (1.f - wy) * wx },
                        { wy * (1.f - wx),         wy * wx } };
    const float* tb = t + b * 128 * 128;
    float r = 0.f, g = 0.f, bl = 0.f;
#pragma unroll
    for (int dy = 0; dy < 2; ++dy)
#pragma unroll
        for (int dx = 0; dx < 2; ++dx) {
            float tv = tb[yy[dy] * 128 + xx[dx]];
            float nrm = (tv - mn) * inv;
            int idx = (int)(nrm * 256.f);
            idx = min(max(idx, 0), 255);
            float w = wgt[dy][dx];
            r  += w * lut[idx * 3 + 0];
            g  += w * lut[idx * 3 + 1];
            bl += w * lut[idx * 3 + 2];
        }
    int base = (b * 3) * 224 * 224 + oy * 224 + ox;
    img[base]                 = r;
    img[base + 224 * 224]     = g;
    img[base + 2 * 224 * 224] = bl;
}

// ---------------------------------------------------------------------------
// 4) Stem im2col: img fp32 NCHW (16,3,224,224) -> A0 bf16 [200704, 160]
//    K order (ky,kx,c), K=147 zero-padded to 160 (multiple of 32).
//    7x7 conv, stride 2, pad 3 -> 112x112 outputs.
// ---------------------------------------------------------------------------
__global__ void im2col_stem_kernel(const float* __restrict__ img,
                                   __bf16* __restrict__ A0, int total) {
    for (int i = blockIdx.x * blockDim.x + threadIdx.x; i < total;
         i += gridDim.x * blockDim.x) {
        int p = i / 160;
        int k = i % 160;
        float v = 0.f;
        if (k < 147) {
            int c   = k % 3;
            int tap = k / 3;
            int ky = tap / 7, kx = tap % 7;
            int b   = p / (112 * 112);
            int rem = p % (112 * 112);
            int oh = rem / 112, ow = rem % 112;
            int ih = oh * 2 - 3 + ky, iw = ow * 2 - 3 + kx;
            if ((unsigned)ih < 224u && (unsigned)iw < 224u)
                v = img[((b * 3 + c) * 224 + ih) * 224 + iw];
        }
        A0[i] = (__bf16)v;
    }
}

// ---------------------------------------------------------------------------
// 5) Weight packing: fp32 OIHW (Cout,Cin,KH,KW) -> bf16 fragment-ready tiles.
//    K order = (ky,kx,c), zero padded to Kpad.  Cout grouped by 64 channels
//    (4 consecutive 16-wide n-tiles).  Block order: ((g*Kt + kt)*4 + q), so a
//    group's whole weight stream is one contiguous run (4 KB per k-tile).
//    Within a 32x16 block (contiguous k order, matching the A fragment):
//      lane l: column n = l&15, k = (l>>4)*16 + e   (e = 0..15, contiguous)
// ---------------------------------------------------------------------------
__global__ void pack_weights_kernel(const float* __restrict__ W,
                                    __bf16* __restrict__ Wp,
                                    int Cout, int Cin, int KH, int KW,
                                    int Kpad, int total) {
    int Kt = Kpad >> 5;
    int Kreal = KH * KW * Cin;
    for (int i = blockIdx.x * blockDim.x + threadIdx.x; i < total;
         i += gridDim.x * blockDim.x) {
        int e    = i & 15;
        int lane = (i >> 4) & 31;
        int blk  = i >> 9;
        int q    = blk & 3;
        int rest = blk >> 2;
        int kt   = rest % Kt;
        int g    = rest / Kt;
        int n = (g * 4 + q) * 16 + (lane & 15);
        int k = kt * 32 + (lane >> 4) * 16 + e;
        float v = 0.f;
        if (k < Kreal) {
            int c   = k % Cin;
            int tap = k / Cin;
            int ky = tap / KW, kx = tap % KW;
            v = W[((n * Cin + c) * KH + ky) * KW + kx];
        }
        Wp[i] = (__bf16)v;
    }
}

// BN fold: scale = g/sqrt(v+eps), bias = b - m*scale
__global__ void bnfold_kernel(const float* __restrict__ g,
                              const float* __restrict__ b,
                              const float* __restrict__ m,
                              const float* __restrict__ v,
                              float* __restrict__ sc, float* __restrict__ bi,
                              int C) {
    int i = blockIdx.x * blockDim.x + threadIdx.x;
    if (i >= C) return;
    float s = g[i] * __frsqrt_rn(v[i] + EPSB);
    sc[i] = s;
    bi[i] = b[i] - m[i] * s;
}

// ---------------------------------------------------------------------------
// 6) Implicit-GEMM WMMA conv, register-blocked 4x in N, with async
//    global->LDS double-buffered weight staging.
//
//    One wave computes a 16(pixels) x 64(out-channels) f32 tile: 4
//    accumulators, each A fragment reused by 4 back-to-back
//    v_wmma_f32_16x16x32_bf16 ops.  All 8 waves of a block share one
//    64-channel group; the group's 4 KB weight k-chunk is staged into LDS
//    once per block via global_load_async_to_lds_b128 (ASYNCcnt), double
//    buffered so the copy of chunk k+1 overlaps the WMMAs of chunk k.
//    B fragments are then read from LDS (ds_load_b128, broadcast-friendly).
//
//    A fragment (16x32 bf16), contiguous k order:
//      lane l: row m = l&15,   k = (l>>4)*16 + {0..15}
//      -> two contiguous 16B loads per lane from the NHWC channel run.
// ---------------------------------------------------------------------------
__global__ void conv_wmma_kernel(const __bf16* __restrict__ X,
                                 const __bf16* __restrict__ Wp,
                                 const float* __restrict__ sc,
                                 const float* __restrict__ bi,
                                 const __bf16* __restrict__ res,
                                 __bf16* __restrict__ Y,
                                 int B, int Hin, int Win, int Cin,
                                 int Hout, int Wout, int Cout,
                                 int KH, int KW, int stride, int pad,
                                 int relu) {
    __shared__ __attribute__((aligned(16))) __bf16 smem[2][2048]; // 2 x 4KB

    const int tid  = threadIdx.x;
    const int lane = tid & 31;
    const int wib  = tid >> 5;                  // wave in block, 0..7
    const int ngrp = Cout >> 6;                 // 64-channel groups
    const unsigned bg = blockIdx.x;
    const int g = (int)(bg % (unsigned)ngrp);
    const unsigned mt = (bg / (unsigned)ngrp) * 8u + (unsigned)wib;
    const unsigned Mt = (unsigned)(B * Hout * Wout) >> 4;
    const bool active = (mt < Mt);

    const int khalf = lane >> 4;
    const unsigned p   = mt * 16u + (unsigned)(lane & 15);
    const unsigned hw  = (unsigned)(Hout * Wout);
    const int b   = (int)(p / hw);
    const int rem = (int)(p % hw);
    const int oh  = rem / Wout, ow = rem % Wout;
    const int ih0 = oh * stride - pad;
    const int iw0 = ow * stride - pad;

    const int KT = (KH * KW * Cin) >> 5;        // total 32-wide k-chunks
    const __bf16* wstream = Wp + (size_t)g * KT * 2048;

    // async stage of one 4KB weight chunk: 256 threads x 16B
    auto stage = [&](int bufi, int kt) {
        unsigned lds = (unsigned)(uintptr_t)(&smem[bufi][0]) + tid * 16;
        const char* src = (const char*)wstream + (size_t)kt * 4096 + tid * 16;
        asm volatile("global_load_async_to_lds_b128 %0, %1, off"
                     :: "v"(lds), "v"(src)
                     : "memory");
    };

    v8f acc[4];
#pragma unroll
    for (int q = 0; q < 4; ++q)
        acc[q] = (v8f){0.f, 0.f, 0.f, 0.f, 0.f, 0.f, 0.f, 0.f};

    // prologue: stage chunk 0
    stage(0, 0);
    wait_async0();
    __syncthreads();

    int buf = 0;
    int ktNext = 1;
    for (int ky = 0; ky < KH; ++ky) {
        int ih = ih0 + ky;
        bool rowok = (unsigned)ih < (unsigned)Hin;
        for (int kx = 0; kx < KW; ++kx) {
            int iw = iw0 + kx;
            bool inb = active && rowok && ((unsigned)iw < (unsigned)Win);
            int base = ((b * Hin + ih) * Win + iw) * Cin + khalf * 16;
            for (int c0 = 0; c0 < Cin; c0 += 32) {
                // overlap: kick off the next chunk's copy
                if (ktNext < KT) stage(buf ^ 1, ktNext);
                ++ktNext;

                Frag16 a;
                if (inb) {
                    a.h[0] = *(const v8bf*)(X + base + c0);
                    a.h[1] = *(const v8bf*)(X + base + c0 + 8);
                    __builtin_prefetch((const void*)(X + base + c0 + 64),
                                       0, 1);
                } else {
#pragma unroll
                    for (int j = 0; j < 8; ++j) a.u[j] = 0u;
                }

                const __bf16* ws = &smem[buf][0];
                Frag16 w[4];
#pragma unroll
                for (int q = 0; q < 4; ++q) {
                    w[q].h[0] = *(const v8bf*)(ws + q * 512 + lane * 16);
                    w[q].h[1] = *(const v8bf*)(ws + q * 512 + lane * 16 + 8);
                }
#pragma unroll
                for (int q = 0; q < 4; ++q)
                    acc[q] = __builtin_amdgcn_wmma_f32_16x16x32_bf16(
                        false, a.v, false, w[q].v, (short)0, acc[q],
                        false, false);

                // next chunk's copy must be complete and everyone done
                // reading before buffers swap
                wait_async0();
                __syncthreads();
                buf ^= 1;
            }
        }
    }

    if (!active) return;

    // Epilogue. C/D layout: VGPR j -> row m = j + 8*(lane>>4), col n = lane&15
    const int nb = (g * 4) * 16 + (lane & 15);
    const int pbase = (int)(mt * 16u);
#pragma unroll
    for (int q = 0; q < 4; ++q) {
        int n = nb + q * 16;
        float s = sc[n];
        float t = bi[n];
#pragma unroll
        for (int j = 0; j < 8; ++j) {
            int m  = j + khalf * 8;
            int pp = pbase + m;
            float v = acc[q][j] * s + t;
            if (res) v += (float)res[pp * Cout + n];
            if (relu && v < 0.f) v = 0.f;
            Y[pp * Cout + n] = (__bf16)v;
        }
    }
}

// ---------------------------------------------------------------------------
// 7) MaxPool 3x3 / stride 2 / pad 1 on NHWC bf16
// ---------------------------------------------------------------------------
__global__ void maxpool_kernel(const __bf16* __restrict__ X,
                               __bf16* __restrict__ Y,
                               int B, int H, int W, int C, int Ho, int Wo) {
    int i = blockIdx.x * blockDim.x + threadIdx.x;
    int total = B * Ho * Wo * C;
    if (i >= total) return;
    int c  = i % C;
    int t2 = i / C;
    int ow = t2 % Wo; t2 /= Wo;
    int oh = t2 % Ho;
    int b  = t2 / Ho;
    float mx = -3.4e38f;
#pragma unroll
    for (int dy = 0; dy < 3; ++dy)
#pragma unroll
        for (int dx = 0; dx < 3; ++dx) {
            int ih = oh * 2 - 1 + dy, iw = ow * 2 - 1 + dx;
            if ((unsigned)ih < (unsigned)H && (unsigned)iw < (unsigned)W) {
                float v = (float)X[((b * H + ih) * W + iw) * C + c];
                mx = fmaxf(mx, v);
            }
        }
    Y[i] = (__bf16)mx;
}

// ---------------------------------------------------------------------------
// 8) Global average pool (16,7,7,2048) bf16 -> (16,2048) f32
// ---------------------------------------------------------------------------
__global__ void avgpool_kernel(const __bf16* __restrict__ X,
                               float* __restrict__ Y, int C, int HW) {
    int i = blockIdx.x * blockDim.x + threadIdx.x;
    if (i >= 16 * C) return;
    int b = i / C, c = i % C;
    float s = 0.f;
    for (int j = 0; j < HW; ++j)
        s += (float)X[(b * HW + j) * C + c];
    Y[i] = s / (float)HW;
}

// ---------------------------------------------------------------------------
// 9) FC: y = x @ W^T + bias, optional BN1d + LeakyReLU(0.01)
// ---------------------------------------------------------------------------
__global__ void fc_kernel(const float* __restrict__ X,
                          const float* __restrict__ W,
                          const float* __restrict__ bias,
                          const float* __restrict__ g,
                          const float* __restrict__ bb,
                          const float* __restrict__ m,
                          const float* __restrict__ v,
                          float* __restrict__ Y, int Bn, int K, int N,
                          int act) {
    int i = blockIdx.x * blockDim.x + threadIdx.x;
    if (i >= Bn * N) return;
    int b = i / N, n = i % N;
    const float* x = X + b * K;
    const float* w = W + n * K;
    float s = 0.f;
    for (int k = 0; k < K; ++k) s += x[k] * w[k];
    s += bias[n];
    if (act) {
        float scl = g[n] * __frsqrt_rn(v[n] + EPSB);
        s = s * scl + (bb[n] - m[n] * scl);
        s = s > 0.f ? s : 0.01f * s;
    }
    Y[i] = s;
}

// ---------------------------------------------------------------------------
// Host orchestration
// ---------------------------------------------------------------------------
struct BNp { const float *g, *b, *m, *v; };

extern "C" void kernel_launch(void* const* d_in, const int* in_sizes, int n_in,
                              void* d_out, int out_size, void* d_ws,
                              size_t ws_size, hipStream_t stream) {
    (void)in_sizes; (void)n_in; (void)out_size; (void)ws_size;

    // ---- walk inputs in setup_inputs() flatten order ----
    int ii = 0;
    auto nextf = [&]() { return (const float*)d_in[ii++]; };
    auto nextBN = [&]() { BNp p; p.g = nextf(); p.b = nextf();
                          p.m = nextf(); p.v = nextf(); return p; };

    const float* mfcc = nextf();
    const float* llfs = nextf();
    const float* lut  = nextf();
    const float* conv1w = nextf();
    BNp bn1 = nextBN();

    static const int PLANES[4] = {64, 128, 256, 512};
    static const int NBLK[4]   = {3, 4, 6, 3};
    static const int STR[4]    = {1, 2, 2, 2};

    struct Blk {
        const float *w1, *w2, *w3, *wd;
        BNp b1, b2, b3, bd;
        int planes, stride, cin, has_d;
    } blks[16];
    int nblk = 0, cin = 64;
    for (int li = 0; li < 4; ++li)
        for (int bi = 0; bi < NBLK[li]; ++bi) {
            Blk& k = blks[nblk++];
            int s = (bi == 0) ? STR[li] : 1;
            k.planes = PLANES[li]; k.stride = s; k.cin = cin;
            k.w1 = nextf(); k.b1 = nextBN();
            k.w2 = nextf(); k.b2 = nextBN();
            k.w3 = nextf(); k.b3 = nextBN();
            k.has_d = (s != 1 || cin != PLANES[li] * 4);
            if (k.has_d) { k.wd = nextf(); k.bd = nextBN(); }
            else         { k.wd = nullptr; }
            cin = PLANES[li] * 4;
        }
    const float* fcw[3]; const float* fcb[3]; BNp fcbn[3];
    for (int i = 0; i < 3; ++i) { fcw[i] = nextf(); fcb[i] = nextf();
                                  fcbn[i] = nextBN(); }
    const float* lastw = nextf();
    const float* lastb = nextf();

    // ---- workspace bump allocator (same layout every call) ----
    size_t off = 0;
    auto alloc = [&](size_t bytes) -> void* {
        void* p = (char*)d_ws + off;
        off += (bytes + 255) & ~(size_t)255;
        return p;
    };
    float*  t     = (float*)alloc((size_t)16 * 128 * 128 * 4);
    float*  part  = (float*)alloc(512 * 4);
    float*  mm    = (float*)alloc(2 * 4);
    float*  img   = (float*)alloc((size_t)16 * 3 * 224 * 224 * 4);
    __bf16* A0    = (__bf16*)alloc((size_t)200704 * 160 * 2);
    const size_t SLOT = (size_t)16 * 112 * 112 * 64 * 2;   // 25.7 MB
    __bf16* slot[4];
    for (int i = 0; i < 4; ++i) slot[i] = (__bf16*)alloc(SLOT);
    float* pooled = (float*)alloc((size_t)16 * 2048 * 4);
    float* fcb1   = (float*)alloc((size_t)16 * 512 * 4);
    float* fcb2   = (float*)alloc((size_t)16 * 512 * 4);

    auto cdiv = [](long a, long b) { return (a + b - 1) / b; };

    // pack weights + fold BN + launch one WMMA conv
    auto conv = [&](const __bf16* X, const float* Wsrc,
                    int srcCin, int srcKH, int srcKW, int Kpad,
                    BNp bn, int B, int Hin, int Win, int gCin,
                    int Cout, int gKH, int gKW, int stride, int pad,
                    int relu, const __bf16* res, __bf16* Y,
                    int Hout, int Wout) {
        __bf16* Wp = (__bf16*)alloc((size_t)Kpad * Cout * 2);
        int wtotal = Kpad * Cout;
        pack_weights_kernel<<<(int)cdiv(wtotal, 256), 256, 0, stream>>>(
            Wsrc, Wp, Cout, srcCin, srcKH, srcKW, Kpad, wtotal);
        float* sc = (float*)alloc((size_t)Cout * 8);
        float* bi = sc + Cout;
        bnfold_kernel<<<(int)cdiv(Cout, 256), 256, 0, stream>>>(
            bn.g, bn.b, bn.m, bn.v, sc, bi, Cout);
        long Mt = ((long)B * Hout * Wout) / 16;
        long blocks = cdiv(Mt, 8) * (Cout / 64);
        conv_wmma_kernel<<<(int)blocks, 256, 0, stream>>>(
            X, Wp, sc, bi, res, Y, B, Hin, Win, gCin, Hout, Wout, Cout,
            gKH, gKW, stride, pad, relu);
    };

    // ---- preprocessing ----
    resize_concat_kernel<<<(16 * 128 * 128 + 255) / 256, 256, 0, stream>>>(
        mfcc, llfs, t);
    minmax_part_kernel<<<256, 256, 0, stream>>>(t, 16 * 128 * 128, part);
    minmax_final_kernel<<<1, 256, 0, stream>>>(part, mm);
    colorize_resize_kernel<<<(16 * 224 * 224 + 255) / 256, 256, 0, stream>>>(
        t, mm, lut, img);

    // ---- stem: im2col (K=147 pad 160) + WMMA GEMM + BN + ReLU ----
    int a0total = 200704 * 160;
    im2col_stem_kernel<<<(int)cdiv(a0total, 256), 256, 0, stream>>>(
        img, A0, a0total);
    conv(A0, conv1w, /*srcCin*/3, /*srcKH*/7, /*srcKW*/7, /*Kpad*/160,
         bn1, 16, 112, 112, /*gCin*/160, 64, 1, 1, 1, 0, /*relu*/1,
         nullptr, slot[0], 112, 112);

    // ---- maxpool 3x3/2 pad1: 112 -> 56 ----
    long mp = (long)16 * 56 * 56 * 64;
    maxpool_kernel<<<(int)cdiv(mp, 256), 256, 0, stream>>>(
        slot[0], slot[1], 16, 112, 112, 64, 56, 56);

    // ---- bottleneck stack ----
    int curi = 1, H = 56, C = 64;
    for (int i = 0; i < nblk; ++i) {
        Blk& k = blks[i];
        int t1 = (curi + 1) & 3, t2 = (curi + 2) & 3, t3 = (curi + 3) & 3;
        int pl = k.planes, s = k.stride;
        int H2 = (H - 1) / s + 1;

        // conv1: 1x1, C -> pl, stride 1, BN+ReLU
        conv(slot[curi], k.w1, C, 1, 1, C, k.b1,
             16, H, H, C, pl, 1, 1, 1, 0, 1, nullptr, slot[t1], H, H);
        // conv2: 3x3, pl -> pl, stride s, pad 1, BN+ReLU
        conv(slot[t1], k.w2, pl, 3, 3, 9 * pl, k.b2,
             16, H, H, pl, pl, 3, 3, s, 1, 1, nullptr, slot[t2], H2, H2);
        // shortcut
        const __bf16* res;
        if (k.has_d) {
            conv(slot[curi], k.wd, C, 1, 1, C, k.bd,
                 16, H, H, C, pl * 4, 1, 1, s, 0, 0, nullptr, slot[t3],
                 H2, H2);
            res = slot[t3];
        } else {
            res = slot[curi];
        }
        // conv3: 1x1, pl -> 4*pl, BN, +residual, ReLU (fused)
        conv(slot[t2], k.w3, pl, 1, 1, pl, k.b3,
             16, H2, H2, pl, pl * 4, 1, 1, 1, 0, 1, res, slot[t1], H2, H2);

        curi = t1; H = H2; C = pl * 4;
    }

    // ---- head ----
    avgpool_kernel<<<(16 * 2048 + 255) / 256, 256, 0, stream>>>(
        slot[curi], pooled, 2048, H * H);   // H == 7
    float* out = (float*)d_out;             // [0,4096) feat, [4096,4272) logits
    fc_kernel<<<(16 * 512 + 255) / 256, 256, 0, stream>>>(
        pooled, fcw[0], fcb[0], fcbn[0].g, fcbn[0].b, fcbn[0].m, fcbn[0].v,
        fcb1, 16, 2048, 512, 1);
    fc_kernel<<<(16 * 512 + 255) / 256, 256, 0, stream>>>(
        fcb1, fcw[1], fcb[1], fcbn[1].g, fcbn[1].b, fcbn[1].m, fcbn[1].v,
        fcb2, 16, 512, 512, 1);
    fc_kernel<<<(16 * 256 + 255) / 256, 256, 0, stream>>>(
        fcb2, fcw[2], fcb[2], fcbn[2].g, fcbn[2].b, fcbn[2].m, fcbn[2].v,
        out, 16, 512, 256, 1);
    fc_kernel<<<(16 * 11 + 255) / 256, 256, 0, stream>>>(
        out, lastw, lastb, nullptr, nullptr, nullptr, nullptr,
        out + 4096, 16, 256, 11, 0);
}
